// Model_74938589381339
// MI455X (gfx1250) — matrix operand order repaired and checked by reading, not compiled, not run on hardware
//
#include <hip/hip_runtime.h>
#include <hip/hip_bf16.h>

typedef __attribute__((ext_vector_type(16))) _Float16 v16h;
typedef __attribute__((ext_vector_type(8)))  _Float16 v8h;
typedef __attribute__((ext_vector_type(8)))  float    v8f;

#define T_STEPS   2048
#define BATCH     64
#define H_DIM     200
#define KPAD      224          // K (=H) padded to 7*32 for wmma 16x16x32
#define KT_TILES  7
#define NT_TILES  52           // 4H=800 packed gate cols -> 832 = 52*16
#define NWG       16           // one (potential) cluster of 16 WGPs
#define WAVES_PER_WG 13
#define THREADS   (WAVES_PER_WG * 32)   // 416
#define TILES_PER_WAVE 21      // 7 (L0 Whh0) + 7 (L1 Wih1) + 7 (L1 Whh1)
#define FRAG_HALVES 16         // per-lane halves of one 16x16(x32) B fragment
#define LDS_HALVES (WAVES_PER_WG * TILES_PER_WAVE * 32 * FRAG_HALVES)
#define LDS_BYTES  (LDS_HALVES * 2)     // 279,552 B of the WGP's 320KB

// ---- workspace layout (bytes) ----
#define OFF_SYNC   0
#define OFF_H1     256
#define HBUF_BYTES (BATCH * KPAD * 2)   // 28,672 (f16)
#define HBUF_HALVES (BATCH * KPAD)
#define OFF_H2     (OFF_H1 + 2 * HBUF_BYTES)
#define OFF_PART   (OFF_H2 + 2 * HBUF_BYTES)
#define PART_BYTES (NT_TILES * BATCH * 4)
#define WS_ZERO_BYTES (OFF_PART + PART_BYTES)

// CDNA5 has a native TRANS32 tanh (v_tanh_f32). Use it when the compiler
// exposes the builtin; otherwise fall back to the exp/rcp formulation.
#if __has_builtin(__builtin_amdgcn_tanhf)
__device__ __forceinline__ float tanh_fast(float x) {
  return __builtin_amdgcn_tanhf(x);
}
#else
__device__ __forceinline__ float tanh_fast(float x) {
  // 2*sigmoid(2x)-1 : saturates cleanly at +-1, no inf/inf NaN
  return 2.0f * __builtin_amdgcn_rcpf(1.0f + __expf(-2.0f * x)) - 1.0f;
}
#endif
__device__ __forceinline__ float sigmoid_fast(float x) {
  // sigmoid(x) = 0.5*tanh(x/2) + 0.5  (one TRANS + one FMA on CDNA5)
  return __builtin_fmaf(0.5f, tanh_fast(0.5f * x), 0.5f);
}

// A-fragment (16x32 f16) load, ISA layout: lane<16 holds K {0..7,16..23} of
// row M=lane; lane>=16 holds K {8..15,24..31} of row M=lane-16.
__device__ __forceinline__ v16h load_a_frag(const _Float16* __restrict__ hbuf,
                                            int mt, int kt, int lane) {
  const int hi  = lane >> 4;
  const int row = mt * 16 + (lane & 15);
  const int k0  = kt * 32 + hi * 8;
  const _Float16* p = hbuf + row * KPAD + k0;
  union { v16h v; v8h h[2]; } u;
  u.h[0] = *(const v8h*)(p);        // K = k0 .. k0+7
  u.h[1] = *(const v8h*)(p + 16);   // K = k0+16 .. k0+23
  return u.v;
}

// One K-chain: preload all A and B fragments, then run the 7-WMMA chain so the
// compiler can batch/pipe the ds & global waits.
__device__ __forceinline__ v8f gemm_chain(v8f c,
                                          const _Float16* __restrict__ abase,
                                          int mt, int lane,
                                          const v16h* lfr, int tile0) {
  v16h a[KT_TILES], b[KT_TILES];
#pragma unroll
  for (int kt = 0; kt < KT_TILES; ++kt) a[kt] = load_a_frag(abase, mt, kt, lane);
#pragma unroll
  for (int kt = 0; kt < KT_TILES; ++kt) b[kt] = lfr[(tile0 + kt) * 32];
#pragma unroll
  for (int kt = 0; kt < KT_TILES; ++kt)
    c = __builtin_amdgcn_wmma_f32_16x16x32_f16(false, a[kt], false, b[kt],
                                               (short)0, c, false, false);
  return c;
}

__device__ __forceinline__ void grid_sync(unsigned* cnt, unsigned* sense,
                                          unsigned epoch, bool use_cluster) {
  __threadfence();
  __syncthreads();
  if (use_cluster) {
    // CDNA5 cluster split-barrier: one wave per WG signals, all waves wait.
    if ((threadIdx.x >> 5) == 0)
      asm volatile("s_barrier_signal -3" ::: "memory");
    asm volatile("s_barrier_wait -3" ::: "memory");
  } else {
    // Epoch-counting L2 barrier (monotonic, no counter reset race).
    if (threadIdx.x == 0) {
      unsigned t = __hip_atomic_fetch_add(cnt, 1u, __ATOMIC_ACQ_REL,
                                          __HIP_MEMORY_SCOPE_AGENT);
      if (t == epoch * NWG + (NWG - 1)) {
        __hip_atomic_fetch_add(sense, 1u, __ATOMIC_ACQ_REL,
                               __HIP_MEMORY_SCOPE_AGENT);
      } else {
        while (__hip_atomic_load(sense, __ATOMIC_ACQUIRE,
                                 __HIP_MEMORY_SCOPE_AGENT) <= epoch)
          __builtin_amdgcn_s_sleep(2);
      }
    }
    __syncthreads();
  }
  __threadfence();
}

extern "C" __global__ void ws_zero_kernel(unsigned* p, int n) {
  int i = blockIdx.x * blockDim.x + threadIdx.x;
  if (i < n) p[i] = 0u;
}

extern "C" __global__ __launch_bounds__(THREADS, 1)
void lstm2_persistent(
    const float* __restrict__ x,
    const float* __restrict__ Wih0, const float* __restrict__ Whh0,
    const float* __restrict__ bih0, const float* __restrict__ bhh0,
    const float* __restrict__ Wih1, const float* __restrict__ Whh1,
    const float* __restrict__ bih1, const float* __restrict__ bhh1,
    const float* __restrict__ Wout, const float* __restrict__ bout,
    float* __restrict__ out, char* __restrict__ ws)
{
  extern __shared__ _Float16 smem[];

  unsigned* syncCnt   = (unsigned*)(ws + OFF_SYNC);
  unsigned* syncSense = syncCnt + 1;
  // Single-base + index-select ping-pong (keeps GLOBAL addrspace inference).
  _Float16* h1base = (_Float16*)(ws + OFF_H1);
  _Float16* h2base = (_Float16*)(ws + OFF_H2);
  float* part = (float*)(ws + OFF_PART);

  const int lane = threadIdx.x & 31;
  const int wv   = threadIdx.x >> 5;
  const int w    = blockIdx.x * WAVES_PER_WG + wv;   // 0..207
  const int mt   = w / NT_TILES;                     // 0..3  (batch tile)
  const int nt   = w % NT_TILES;                     // 0..51 (gate-col tile)
  const int hi   = lane >> 4;
  const int nl   = lane & 15;
  const int jj   = nt * 4 + (nl >> 2);               // hidden-unit index
  const int gg   = nl & 3;                           // gate: 0=i 1=f 2=g 3=o
  const bool valid = (jj < H_DIM);
  const int orow = gg * H_DIM + jj;                  // row in original [4H,H] W

  // In-cluster detection: IB_STS2.CLUSTER_ID (bits 9:6), 0 => not clustered.
  unsigned clid;
  asm volatile("s_getreg_b32 %0, hwreg(HW_REG_IB_STS2, 6, 4)" : "=s"(clid));
  const bool use_cluster = (clid != 0);

  // ---- one-time: pack this wave's 21 B-matrix fragments (f32 -> f16) into LDS.
  // B layout (32x16 f16): lane<16 holds K {0..7,16..23} of col N=lane,
  // lane>=16 holds K {8..15,24..31} of col N=lane-16 (matches A).
  {
    const float* wsrc[3] = { Whh0, Wih1, Whh1 };
    for (int tile = 0; tile < TILES_PER_WAVE; ++tile) {
      const int m  = tile / KT_TILES;
      const int kt = tile % KT_TILES;
      const int k0 = kt * 32 + hi * 8;
      const float* Wm = wsrc[m];
      union { v16h v; _Float16 e[16]; } u;
#pragma unroll
      for (int q = 0; q < 16; ++q) {
        const int k = k0 + q + (q & 8);
        float val = (valid && k < H_DIM) ? Wm[orow * H_DIM + k] : 0.0f;
        u.e[q] = (_Float16)val;
      }
      *((v16h*)smem + (wv * TILES_PER_WAVE + tile) * 32 + lane) = u.v;
    }
  }
  __syncthreads();

  // Per-lane constants.
  const float wih0v = valid ? Wih0[orow] : 0.0f;              // W_ih0 is [800,1]
  const float bias0 = valid ? (bih0[orow] + bhh0[orow]) : 0.0f;
  const float bias1 = valid ? (bih1[orow] + bhh1[orow]) : 0.0f;
  const float woutv = valid ? Wout[jj] : 0.0f;
  const float bo    = bout[0];
  const int   gbase = lane & ~3;

  const v16h* lfr = (const v16h*)smem + wv * TILES_PER_WAVE * 32 + lane;

  // Cell state lives in registers for the whole sequence.
  float c1s[8], c2s[8];
#pragma unroll
  for (int r = 0; r < 8; ++r) { c1s[r] = 0.0f; c2s[r] = 0.0f; }

  unsigned epoch = 0;
  int pp = 0;                       // ping-pong selector (prev buffers)
#pragma unroll 1
  for (int t = 0; t < T_STEPS; ++t) {
    const int cur = pp ^ 1;
    const _Float16* h1prev = h1base + pp  * HBUF_HALVES;
    _Float16*       h1cur  = h1base + cur * HBUF_HALVES;
    const _Float16* h2prev = h2base + pp  * HBUF_HALVES;
    _Float16*       h2cur  = h2base + cur * HBUF_HALVES;

    // ================= layer 0: gates0 = x_t*Wih0 + b + h1_prev @ Whh0^T
    v8f c;
#pragma unroll
    for (int r = 0; r < 8; ++r) {
      float xv = x[t * BATCH + mt * 16 + hi * 8 + r];
      c[r] = xv * wih0v + bias0;
    }
    c = gemm_chain(c, h1prev, mt, lane, lfr, 0 * KT_TILES);

#pragma unroll
    for (int r = 0; r < 8; ++r) {
      float pre = c[r];
      float vi = __shfl(pre, gbase + 0, 32);
      float vf = __shfl(pre, gbase + 1, 32);
      float vg = __shfl(pre, gbase + 2, 32);
      float vo = __shfl(pre, gbase + 3, 32);
      float cc = sigmoid_fast(vf) * c1s[r] + sigmoid_fast(vi) * tanh_fast(vg);
      c1s[r] = cc;
      float h = sigmoid_fast(vo) * tanh_fast(cc);
      if (gg == 0 && valid)
        h1cur[(mt * 16 + hi * 8 + r) * KPAD + jj] = (_Float16)h;
    }
    grid_sync(syncCnt, syncSense, epoch++, use_cluster);

    // ================= layer 1: gates1 = b + h1_t @ Wih1^T + h2_prev @ Whh1^T
#pragma unroll
    for (int r = 0; r < 8; ++r) c[r] = bias1;
    c = gemm_chain(c, h1cur,  mt, lane, lfr, 1 * KT_TILES);
    c = gemm_chain(c, h2prev, mt, lane, lfr, 2 * KT_TILES);

#pragma unroll
    for (int r = 0; r < 8; ++r) {
      float pre = c[r];
      float vi = __shfl(pre, gbase + 0, 32);
      float vf = __shfl(pre, gbase + 1, 32);
      float vg = __shfl(pre, gbase + 2, 32);
      float vo = __shfl(pre, gbase + 3, 32);
      float cc = sigmoid_fast(vf) * c2s[r] + sigmoid_fast(vi) * tanh_fast(vg);
      c2s[r] = cc;
      float h = sigmoid_fast(vo) * tanh_fast(cc);
      const int brow = mt * 16 + hi * 8 + r;
      if (gg == 0 && valid)
        h2cur[brow * KPAD + jj] = (_Float16)h;
      // output-head partial: sum over this wave's 4 hidden units
      float p = h * woutv;                 // identical across the 4 gate lanes
      p += __shfl_xor(p, 4, 32);
      p += __shfl_xor(p, 8, 32);
      if (nl == 0) part[nt * BATCH + brow] = p;
    }
    grid_sync(syncCnt, syncSense, epoch++, use_cluster);

    // ================= output head: out[t,b] = b_out + sum_nt part[nt][b]
    if (blockIdx.x == 0 && threadIdx.x < BATCH) {
      float s = bo;
#pragma unroll 4
      for (int q = 0; q < NT_TILES; ++q) s += part[q * BATCH + threadIdx.x];
      out[t * BATCH + threadIdx.x] = s;
    }
    pp ^= 1;
  }
}

extern "C" void kernel_launch(void* const* d_in, const int* in_sizes, int n_in,
                              void* d_out, int out_size, void* d_ws, size_t ws_size,
                              hipStream_t stream) {
  (void)in_sizes; (void)n_in; (void)out_size; (void)ws_size;
  const float* x    = (const float*)d_in[0];
  const float* Wih0 = (const float*)d_in[1];
  const float* Whh0 = (const float*)d_in[2];
  const float* bih0 = (const float*)d_in[3];
  const float* bhh0 = (const float*)d_in[4];
  const float* Wih1 = (const float*)d_in[5];
  const float* Whh1 = (const float*)d_in[6];
  const float* bih1 = (const float*)d_in[7];
  const float* bhh1 = (const float*)d_in[8];
  const float* Wout = (const float*)d_in[9];
  const float* bout = (const float*)d_in[10];
  float* out = (float*)d_out;
  char* ws = (char*)d_ws;

  // Zero sync counters, h ping-pong buffers (also provides h_0 = 0; c lives
  // in registers) and the partial-sum scratch.
  const int zwords = WS_ZERO_BYTES / 4;
  ws_zero_kernel<<<(zwords + 255) / 256, 256, 0, stream>>>((unsigned*)d_ws, zwords);

  (void)hipFuncSetAttribute((const void*)lstm2_persistent,
                            hipFuncAttributeMaxDynamicSharedMemorySize,
                            LDS_BYTES);
  lstm2_persistent<<<dim3(NWG), dim3(THREADS), LDS_BYTES, stream>>>(
      x, Wih0, Whh0, bih0, bhh0, Wih1, Whh1, bih1, bhh1, Wout, bout, out, ws);
}
